// NNMFMixerEncoder_35725537968574
// MI455X (gfx1250) — matrix operand order repaired and checked by reading, not compiled, not run on hardware
//
#include <hip/hip_runtime.h>
#include <hip/hip_bf16.h>

// ---------------------------------------------------------------------------
// Types for CDNA5 WMMA (gfx1250, wave32)
// ---------------------------------------------------------------------------
typedef __attribute__((ext_vector_type(16))) __bf16 v16bf;
typedef __attribute__((ext_vector_type(8)))  float  v8f;

union Frag { v16bf v; uint4 u[2]; };

__device__ __forceinline__ v8f wmma_bf16x32(const Frag& a, const Frag& b, v8f c) {
  // D = A(16x32) * B(32x16) + C, f32 accumulate
  return __builtin_amdgcn_wmma_f32_16x16x32_bf16(false, a.v, false, b.v,
                                                 (short)0, c, false, false);
}

__device__ __forceinline__ unsigned short f2bf(float x) {
  union { float f; unsigned int u; } c; c.f = x;
  unsigned int r = c.u + 0x7FFFu + ((c.u >> 16) & 1u);   // round-to-nearest-even
  return (unsigned short)(r >> 16);
}

// Problem constants
#define BB   256
#define SS   196
#define EE   384
#define HH   8
#define EH   48
#define SP   208   // S padded to 16
#define KP   224   // S padded to 32 (WMMA K step)
#define NIT  10
#define NTHR 416   // 13 waves: one M-tile (of 13) per wave in the NNMF GEMMs

// ---------------------------------------------------------------------------
// f32 -> bf16 convert (grid-stride)
// ---------------------------------------------------------------------------
__global__ void cvt_f32_bf16_kernel(const float* __restrict__ s,
                                    unsigned short* __restrict__ d, int n) {
  for (int i = blockIdx.x * blockDim.x + threadIdx.x; i < n;
       i += gridDim.x * blockDim.x)
    d[i] = f2bf(s[i]);
}

// ---------------------------------------------------------------------------
// Build padded bf16 copies of global_weight:  gwb[h][o][i], gwtb[h][i][o]
// both [8][208][224], zero padded
// ---------------------------------------------------------------------------
__global__ void prep_gw_kernel(const float* __restrict__ gw,
                               unsigned short* __restrict__ gwb,
                               unsigned short* __restrict__ gwtb) {
  const int n = HH * SP * KP;
  for (int idx = blockIdx.x * blockDim.x + threadIdx.x; idx < n;
       idx += gridDim.x * blockDim.x) {
    int h = idx / (SP * KP);
    int rem = idx - h * (SP * KP);
    int a = rem / KP;          // row   (<208)
    int c = rem - a * KP;      // col   (<224)
    float v1 = (a < SS && c < SS) ? gw[(size_t)h * SS * SS + a * SS + c] : 0.f;
    float v2 = (a < SS && c < SS) ? gw[(size_t)h * SS * SS + c * SS + a] : 0.f;
    gwb[idx]  = f2bf(v1);      // gw[h][o=a][i=c]
    gwtb[idx] = f2bf(v2);      // gw^T: [h][i=a][o=c] = gw[h][c][a]
  }
}

// ---------------------------------------------------------------------------
// Embed:  inp = l1_48( clip( x @ embed_w^T + b, 1e-6 ) )
// Block: 128 rows x 48 cols (one head chunk). 8 waves, 1 M-subtile each,
// A fragment shared across 3 N-tiles (3 live accumulators).
// ---------------------------------------------------------------------------
__global__ __launch_bounds__(256) void embed_kernel(
    const unsigned short* __restrict__ xb,    // [50176][384] bf16
    const unsigned short* __restrict__ wb,    // [384][384]  bf16 (embed_w)
    const float* __restrict__ bias,           // [384]
    float* __restrict__ inp)                  // [50176][384]
{
  __shared__ float xs[128 * 48];
  __shared__ float rinv[128];
  const int tid = threadIdx.x, lane = tid & 31, wave = tid >> 5;
  const int r = lane & 15, half = lane >> 4;
  const int Mbase = blockIdx.x * 128;
  const int Nbase = blockIdx.y * 48;       // head-aligned (48)

  const unsigned short* arow = xb + (size_t)(Mbase + wave * 16 + r) * 384;
  const unsigned short* b0 = wb + (size_t)(Nbase + r) * 384 + 8 * half;
  const unsigned short* b1 = wb + (size_t)(Nbase + 16 + r) * 384 + 8 * half;
  const unsigned short* b2 = wb + (size_t)(Nbase + 32 + r) * 384 + 8 * half;
  v8f c0 = {0.f,0.f,0.f,0.f,0.f,0.f,0.f,0.f}, c1 = c0, c2 = c0;
#pragma unroll
  for (int ks = 0; ks < 12; ++ks) {
    Frag a, b;
    a.u[0] = *(const uint4*)(arow + ks * 32 + 8 * half);
    a.u[1] = *(const uint4*)(arow + ks * 32 + 8 * half + 16);
    b.u[0] = *(const uint4*)(b0 + ks * 32);
    b.u[1] = *(const uint4*)(b0 + ks * 32 + 16);
    c0 = wmma_bf16x32(a, b, c0);
    b.u[0] = *(const uint4*)(b1 + ks * 32);
    b.u[1] = *(const uint4*)(b1 + ks * 32 + 16);
    c1 = wmma_bf16x32(a, b, c1);
    b.u[0] = *(const uint4*)(b2 + ks * 32);
    b.u[1] = *(const uint4*)(b2 + ks * 32 + 16);
    c2 = wmma_bf16x32(a, b, c2);
  }
  const float bv0 = bias[Nbase + r];
  const float bv1 = bias[Nbase + 16 + r];
  const float bv2 = bias[Nbase + 32 + r];
#pragma unroll
  for (int q = 0; q < 8; ++q) {
    int m = wave * 16 + q + 8 * half;
    xs[m * 48 + r]      = fmaxf(c0[q] + bv0, 1e-6f);
    xs[m * 48 + 16 + r] = fmaxf(c1[q] + bv1, 1e-6f);
    xs[m * 48 + 32 + r] = fmaxf(c2[q] + bv2, 1e-6f);
  }
  __syncthreads();
  if (tid < 128) {
    float s = 0.f;
#pragma unroll
    for (int e = 0; e < 48; ++e) s += xs[tid * 48 + e];
    rinv[tid] = 1.f / s;                     // values >= 1e-6, sum safe
  }
  __syncthreads();
  for (int idx = tid; idx < 128 * 48; idx += 256) {
    int i = idx / 48, e = idx - i * 48;
    inp[(size_t)(Mbase + i) * 384 + Nbase + e] = xs[idx] * rinv[i];
  }
}

// ---------------------------------------------------------------------------
// WMMA GEMM core: out[196][48] = A(global bf16 [208][224]) @ B(LDS bf16^T [48][224])
// 13 waves, one 16-row M-tile each; A fragment shared across the 3 N-tiles.
// ---------------------------------------------------------------------------
__device__ __forceinline__ void gemm_196(
    const unsigned short* __restrict__ Ag,   // [208][224] row-major, bf16
    const unsigned short* bT,                // LDS [48][224] (col,k)
    float* outS,                             // LDS [196][48]
    int lane, int wave)
{
  const int r = lane & 15;
  const int half = lane >> 4;
  const int mt = wave;                       // 0..12, uniform per wave
  const unsigned short* arow = Ag + (size_t)(mt * 16 + r) * KP;
  __builtin_prefetch(arow, 0, 0);            // global_prefetch_b8 (L2 stream)
  const unsigned short* b0 = bT + (r)      * KP + 8 * half;
  const unsigned short* b1 = bT + (16 + r) * KP + 8 * half;
  const unsigned short* b2 = bT + (32 + r) * KP + 8 * half;
  v8f c0 = {0.f,0.f,0.f,0.f,0.f,0.f,0.f,0.f}, c1 = c0, c2 = c0;
#pragma unroll
  for (int ks = 0; ks < 7; ++ks) {
    Frag a, b;
    a.u[0] = *(const uint4*)(arow + ks * 32 + 8 * half);
    a.u[1] = *(const uint4*)(arow + ks * 32 + 8 * half + 16);
    b.u[0] = *(const uint4*)(b0 + ks * 32);
    b.u[1] = *(const uint4*)(b0 + ks * 32 + 16);
    c0 = wmma_bf16x32(a, b, c0);
    b.u[0] = *(const uint4*)(b1 + ks * 32);
    b.u[1] = *(const uint4*)(b1 + ks * 32 + 16);
    c1 = wmma_bf16x32(a, b, c1);
    b.u[0] = *(const uint4*)(b2 + ks * 32);
    b.u[1] = *(const uint4*)(b2 + ks * 32 + 16);
    c2 = wmma_bf16x32(a, b, c2);
  }
#pragma unroll
  for (int q = 0; q < 8; ++q) {
    int m = mt * 16 + q + 8 * half;
    if (m < SS) {
      outS[m * 48 + r]      = c0[q];
      outS[m * 48 + 16 + r] = c1[q];
      outS[m * 48 + 32 + r] = c2[q];
    }
  }
}

// ---------------------------------------------------------------------------
// NNMF fixed-point mega-kernel: one block per (batch, head), everything in LDS
// ---------------------------------------------------------------------------
__global__ __launch_bounds__(NTHR) void nnmf_kernel(
    const float* __restrict__ inp_g,          // [B,196,384] (head chunks of 48)
    const float* __restrict__ h0_g,           // [B,196,384]
    const float* __restrict__ lw_g,           // [48,48]
    const unsigned short* __restrict__ gwb,   // [8][208][224]
    const unsigned short* __restrict__ gwtb,  // [8][208][224]
    unsigned short* __restrict__ hf)          // [B,196,384] bf16 out
{
  extern __shared__ char smem[];
  float* inp_s = (float*)smem;                         // [196*48]
  float* h_s   = inp_s + SS * EH;                      // [196*48]
  float* sA    = h_s + SS * EH;                        // [196*48] t1 / ratio
  float* sB    = sA + SS * EH;                         // [196*48] rec / fwd
  unsigned short* bT = (unsigned short*)(sB + SS * EH);// [48*224] h^T / u^T bf16
  float* lw_s  = (float*)(bT + EH * KP);               // [48*48]

  const int tid = threadIdx.x, lane = tid & 31, wave = tid >> 5;
  const int b = blockIdx.x >> 3, head = blockIdx.x & 7;

  const float* ig = inp_g + (size_t)b * SS * EE + head * EH;
  const float* hg = h0_g + (size_t)b * SS * EE + head * EH;
  const unsigned short* Ag_fwd = gwb  + (size_t)head * SP * KP;  // gw  (GEMM2)
  const unsigned short* Ag_rec = gwtb + (size_t)head * SP * KP;  // gw^T (GEMM1)

  for (int idx = tid; idx < EH * EH; idx += NTHR) lw_s[idx] = lw_g[idx];
  for (int idx = tid; idx < EH * KP; idx += NTHR) bT[idx] = 0;   // K-pad once
  for (int idx = tid; idx < SS * EH; idx += NTHR) {
    int i = idx / EH, e = idx - i * EH;
    inp_s[idx] = ig[(size_t)i * EE + e];
    h_s[idx]   = hg[(size_t)i * EE + e];
  }
  __syncthreads();
  // h = l1(h0) per row of 48
  if (tid < SS) {
    float s = 0.f;
#pragma unroll
    for (int e = 0; e < EH; ++e) s += fabsf(h_s[tid * EH + e]);
    float inv = 1.f / fmaxf(s, 1e-12f);
#pragma unroll
    for (int e = 0; e < EH; ++e) h_s[tid * EH + e] *= inv;
  }
  __syncthreads();

  for (int it = 0; it < NIT; ++it) {
    // P1: bT = h^T (bf16); pad region already zero
    for (int idx = tid; idx < EH * SS; idx += NTHR) {
      int f = idx / SS, o = idx - f * SS;
      bT[f * KP + o] = f2bf(h_s[o * EH + f]);
    }
    __syncthreads();
    // GEMM1: t1 = gw^T @ h  -> sA
    gemm_196(Ag_rec, bT, sA, lane, wave);
    __syncthreads();
    // P2: rec = clip(t1 @ lw, 1e-6); ratio = inp * rowsum(rec) / rec -> sA
    if (tid < SS) {
      float row[48];
#pragma unroll
      for (int f = 0; f < 48; ++f) row[f] = sA[tid * EH + f];
      float rsum = 0.f;
      for (int e = 0; e < 48; ++e) {
        float s = 0.f;
#pragma unroll
        for (int f = 0; f < 48; ++f) s += row[f] * lw_s[f * 48 + e];
        s = fmaxf(s, 1e-6f);
        sB[tid * EH + e] = s;
        rsum += s;
      }
      for (int e = 0; e < 48; ++e)
        sA[tid * EH + e] = inp_s[tid * EH + e] * rsum / sB[tid * EH + e];
    }
    __syncthreads();
    // P5: u = ratio @ lw^T -> bT (transposed, bf16); pad region stays zero
    if (tid < SS) {
      float row[48];
#pragma unroll
      for (int e = 0; e < 48; ++e) row[e] = sA[tid * EH + e];
      for (int f = 0; f < 48; ++f) {
        float s = 0.f;
#pragma unroll
        for (int e = 0; e < 48; ++e) s += row[e] * lw_s[f * 48 + e];
        bT[f * KP + tid] = f2bf(s);
      }
    }
    __syncthreads();
    // GEMM2: fwd = gw @ u -> sB
    gemm_196(Ag_fwd, bT, sB, lane, wave);
    __syncthreads();
    // P6: h = l1( clip(h * fwd, 1e-6) )
    if (tid < SS) {
      float s = 0.f;
#pragma unroll
      for (int e = 0; e < 48; ++e) {
        float v = fmaxf(h_s[tid * EH + e] * sB[tid * EH + e], 1e-6f);
        h_s[tid * EH + e] = v;
        s += v;
      }
      float inv = 1.f / s;
#pragma unroll
      for (int e = 0; e < 48; ++e) h_s[tid * EH + e] *= inv;
    }
    __syncthreads();
  }
  // write h (bf16) for the output projection
  for (int idx = tid; idx < SS * EH; idx += NTHR) {
    int i = idx / EH, e = idx - i * EH;
    hf[(size_t)b * SS * EE + (size_t)i * EE + head * EH + e] = f2bf(h_s[idx]);
  }
}

// ---------------------------------------------------------------------------
// Output projection:  out = hf @ out_w^T + out_b
// Same tiling as embed (A shared over 3 N-tiles), direct stores.
// ---------------------------------------------------------------------------
__global__ __launch_bounds__(256) void outproj_kernel(
    const unsigned short* __restrict__ hf,   // [50176][384] bf16
    const unsigned short* __restrict__ wb,   // [384][384]  bf16 (out_w)
    const float* __restrict__ bias,          // [384]
    float* __restrict__ out)                 // [50176][384] f32
{
  const int lane = threadIdx.x & 31, wave = threadIdx.x >> 5;
  const int r = lane & 15, half = lane >> 4;
  const int Mbase = blockIdx.x * 128;
  const int Nbase = blockIdx.y * 48;
  const unsigned short* arow = hf + (size_t)(Mbase + wave * 16 + r) * 384;
  const unsigned short* b0 = wb + (size_t)(Nbase + r) * 384 + 8 * half;
  const unsigned short* b1 = wb + (size_t)(Nbase + 16 + r) * 384 + 8 * half;
  const unsigned short* b2 = wb + (size_t)(Nbase + 32 + r) * 384 + 8 * half;
  v8f c0 = {0.f,0.f,0.f,0.f,0.f,0.f,0.f,0.f}, c1 = c0, c2 = c0;
#pragma unroll
  for (int ks = 0; ks < 12; ++ks) {
    Frag a, b;
    a.u[0] = *(const uint4*)(arow + ks * 32 + 8 * half);
    a.u[1] = *(const uint4*)(arow + ks * 32 + 8 * half + 16);
    b.u[0] = *(const uint4*)(b0 + ks * 32);
    b.u[1] = *(const uint4*)(b0 + ks * 32 + 16);
    c0 = wmma_bf16x32(a, b, c0);
    b.u[0] = *(const uint4*)(b1 + ks * 32);
    b.u[1] = *(const uint4*)(b1 + ks * 32 + 16);
    c1 = wmma_bf16x32(a, b, c1);
    b.u[0] = *(const uint4*)(b2 + ks * 32);
    b.u[1] = *(const uint4*)(b2 + ks * 32 + 16);
    c2 = wmma_bf16x32(a, b, c2);
  }
  const float bv0 = bias[Nbase + r];
  const float bv1 = bias[Nbase + 16 + r];
  const float bv2 = bias[Nbase + 32 + r];
#pragma unroll
  for (int q = 0; q < 8; ++q) {
    size_t m = (size_t)(Mbase + wave * 16 + q + 8 * half);
    out[m * 384 + Nbase + r]      = c0[q] + bv0;
    out[m * 384 + Nbase + 16 + r] = c1[q] + bv1;
    out[m * 384 + Nbase + 32 + r] = c2[q] + bv2;
  }
}

// ---------------------------------------------------------------------------
// Launch
// ---------------------------------------------------------------------------
extern "C" void kernel_launch(void* const* d_in, const int* in_sizes, int n_in,
                              void* d_out, int out_size, void* d_ws, size_t ws_size,
                              hipStream_t stream) {
  const float* x   = (const float*)d_in[0];
  const float* h0  = (const float*)d_in[1];
  const float* ew  = (const float*)d_in[2];
  const float* eb  = (const float*)d_in[3];
  const float* lw  = (const float*)d_in[4];
  const float* gw  = (const float*)d_in[5];
  const float* ow  = (const float*)d_in[6];
  const float* ob  = (const float*)d_in[7];
  float* out = (float*)d_out;

  char* w = (char*)d_ws;
  const size_t nBS = (size_t)BB * SS * EE;                 // 19,267,584
  unsigned short* xb   = (unsigned short*)w;               // x bf16; later reused as hf
  unsigned short* ewb  = (unsigned short*)(w + nBS * 2);
  unsigned short* owb  = ewb + 384 * 384;
  unsigned short* gwb  = owb + 384 * 384;
  unsigned short* gwtb = gwb + HH * SP * KP;
  float* inp = (float*)(gwtb + HH * SP * KP);              // [B,196,384] f32

  cvt_f32_bf16_kernel<<<4096, 256, 0, stream>>>(x, xb, (int)nBS);
  cvt_f32_bf16_kernel<<<576, 256, 0, stream>>>(ew, ewb, 384 * 384);
  cvt_f32_bf16_kernel<<<576, 256, 0, stream>>>(ow, owb, 384 * 384);
  prep_gw_kernel<<<1456, 256, 0, stream>>>(gw, gwb, gwtb);

  embed_kernel<<<dim3(392, 8, 1), 256, 0, stream>>>(xb, ewb, eb, inp);

  // LDS: 4 f32 tiles (196*48) + bf16 transpose buf (48*224) + lw (48*48)
  const size_t lds = (size_t)4 * SS * EH * 4 + (size_t)EH * KP * 2 + 48 * 48 * 4;
  nnmf_kernel<<<BB * HH, NTHR, lds, stream>>>(inp, h0, lw, gwb, gwtb, /*hf=*/xb);

  outproj_kernel<<<dim3(392, 8, 1), 256, 0, stream>>>(xb, owb, ob, out);
}
